// CoverageAttention_2379411882255
// MI455X (gfx1250) — compile-verified, matrix operand order checked by reading, not run on hardware
//
#include <hip/hip_runtime.h>
#include <hip/hip_bf16.h>

// ---------------------------------------------------------------------------
// CoverageAttention fused WMMA + TDM implementation for gfx1250 (MI455X)
//
// dims
#define BSZ 16
#define CCH 684
#define HH  32
#define WW  128
#define HWN 4096       // H*W
#define QCH 256
#define KK  11
#define K2  121        // 11*11
#define NN  256
#define NP  512
#define TT  4
#define KSTEPS1 22     // C=684 padded to 22 * 32
#define KSTEPS2 4      // K2=121 padded to 4 * 32
#define NTILES 32      // NP / 16

typedef __attribute__((ext_vector_type(16))) __bf16        v16bf;
typedef __attribute__((ext_vector_type(8)))  float         v8f;
typedef __attribute__((ext_vector_type(4)))  unsigned int  u32x4v;
typedef __attribute__((ext_vector_type(8)))  int           i32x8v;
typedef __attribute__((ext_vector_type(4)))  int           i32x4v;

union ABregs { u32x4v u[2]; v16bf v; };  // 32 bytes: 16 bf16 = one WMMA operand/lane

#if defined(__gfx1250__) && __has_builtin(__builtin_amdgcn_tensor_load_to_lds)
#define HAVE_TDM 1
#else
#define HAVE_TDM 0
#endif

#if defined(__gfx1250__) && __has_builtin(__builtin_amdgcn_s_wait_tensorcnt)
#define WAIT_TENSORCNT(n) __builtin_amdgcn_s_wait_tensorcnt((short)(n))
#else
#define WAIT_TENSORCNT(n) do {} while (0)
#endif

static __device__ __forceinline__ unsigned int pack_bf16(float a, float b) {
    unsigned int ua = __float_as_uint(a);
    unsigned int ub = __float_as_uint(b);
    unsigned int ra = (ua + 0x7FFFu + ((ua >> 16) & 1u)) >> 16;   // RNE
    unsigned int rb = (ub + 0x7FFFu + ((ub >> 16) & 1u)) >> 16;
    return (ra & 0xFFFFu) | (rb << 16);
}

// Map an even k-offset (0..30) inside a 32-wide k-chunk to the packed
// (lane-half, dword) slot of the 16-bit WMMA A/B operand layout (ISA 7.12.2).
static __device__ __forceinline__ void kslot(int kk, int& half, int& d) {
    half = ((kk & 15) >= 8) ? 1 : 0;
    int kkh = kk - (half ? 8 : 0);           // {0,2,4,6} or {16,18,20,22}
    d = (kkh < 8) ? (kkh >> 1) : (4 + ((kkh - 16) >> 1));
}

#if HAVE_TDM
// Issue a TDM load of a 2D fp32 tile [tile_d1 rows x tile_d0 cols] into LDS.
// D# layout per CDNA5 ISA 8.3/8.4 (2D: groups 2/3 zero, count=1, type=2).
static __device__ __forceinline__ void tdm_load_2d(unsigned lds_addr,
                                                   const float* gptr,
                                                   unsigned tensor_d0,
                                                   unsigned tensor_d1,
                                                   unsigned tile_d0,
                                                   unsigned tile_d1,
                                                   unsigned stride0) {
    unsigned long long ga = (unsigned long long)(size_t)gptr;
    u32x4v g0;
    g0.x = 1u;                                                  // count=1 (user)
    g0.y = lds_addr;                                            // LDS byte addr
    g0.z = (unsigned)(ga & 0xFFFFFFFFull);                      // global_addr lo
    g0.w = (unsigned)((ga >> 32) & 0x1FFFFFFull) | (2u << 30);  // addr hi | type=2
    i32x8v g1;
    g1[0] = (int)(2u << 16);                                    // data_size = 4B
    g1[1] = (int)(tensor_d0 << 16);                             // tensor_dim0 lo
    g1[2] = (int)((tensor_d0 >> 16) | (tensor_d1 << 16));       // dim0 hi|dim1 lo
    g1[3] = (int)((tensor_d1 >> 16) | (tile_d0 << 16));         // dim1 hi|tile0
    g1[4] = (int)(tile_d1 & 0xFFFFu);                           // tile1 | tile2=0
    g1[5] = (int)stride0;                                       // dim0_stride lo
    g1[6] = 0;                                                  // stride hi bits
    g1[7] = 0;
    i32x4v z4 = {0, 0, 0, 0};
#if __clang_major__ >= 23
    i32x8v z8 = {0, 0, 0, 0, 0, 0, 0, 0};
    __builtin_amdgcn_tensor_load_to_lds(g0, g1, z4, z4, z8, 0);
#else
    __builtin_amdgcn_tensor_load_to_lds(g0, g1, z4, z4, 0);
#endif
}
#endif

// ---------------------------------------------------------------------------
// beta[b,hw] = sum_t alpha[b,t,hw]
__global__ void ca_beta_kernel(const float* __restrict__ alpha, float* __restrict__ beta) {
    int idx = blockIdx.x * blockDim.x + threadIdx.x;       // 65536 threads
    int b = idx >> 12, hw = idx & (HWN - 1);
    float s = 0.f;
    #pragma unroll
    for (int t = 0; t < TT; ++t) s += alpha[(b * TT + t) * HWN + hw];
    beta[idx] = s;
}

// ---------------------------------------------------------------------------
// Weff[p,k121] = sum_q Uf_w[p,q] * convQ_w[q,k121]   (fold conv_Q into fc_Uf)
__global__ void ca_weff_kernel(const float* __restrict__ Uf_w,
                               const float* __restrict__ convQ_w,
                               float* __restrict__ Weff) {
    int idx = blockIdx.x * blockDim.x + threadIdx.x;       // 512*121 threads
    int p = idx / K2, k = idx - p * K2;
    float s = 0.f;
    for (int q = 0; q < QCH; ++q) s += Uf_w[p * QCH + q] * convQ_w[q * K2 + k];
    Weff[idx] = s;
}

// ---------------------------------------------------------------------------
// qbias[b,p] = st_hat[b]·Wa_w[p] + Wa_b[p] + convUa_b[p] + Uf_b[p]
//            + sum_q Uf_w[p,q]*convQ_b[q]        (all additive p-biases folded)
__global__ void ca_qbias_kernel(const float* __restrict__ st_hat,
                                const float* __restrict__ Wa_w,
                                const float* __restrict__ Wa_b,
                                const float* __restrict__ convUa_b,
                                const float* __restrict__ Uf_w,
                                const float* __restrict__ Uf_b,
                                const float* __restrict__ convQ_b,
                                float* __restrict__ qbias) {
    int idx = blockIdx.x * blockDim.x + threadIdx.x;       // 16*512 threads
    int b = idx >> 9, p = idx & (NP - 1);
    float s = Wa_b[p] + convUa_b[p] + Uf_b[p];
    for (int n = 0; n < NN; ++n) s += st_hat[b * NN + n] * Wa_w[p * NN + n];
    for (int q = 0; q < QCH; ++q) s += Uf_w[p * QCH + q] * convQ_b[q];
    qbias[idx] = s;
}

// ---------------------------------------------------------------------------
// Pre-pack Ua (convUa_w[p,c]) into bf16 WMMA B-operand lane layout:
// packB[((kstep*NTILES + ntile)*32 + lane)*8 + dword]
__global__ void ca_packB1_kernel(const float* __restrict__ Ua, unsigned int* __restrict__ packB) {
    int g = blockIdx.x * blockDim.x + threadIdx.x;   // 22*32*256 threads
    int d  = g & 7;
    int l  = (g >> 3) & 31;
    int nt = (g >> 8) & 31;
    int ks = g >> 13;
    int n = l & 15;
    int half = (l >= 16) ? 8 : 0;
    int kk = ((d < 4) ? (2 * d) : (16 + 2 * (d - 4))) + half;
    int k0 = ks * 32 + kk;
    int p = nt * 16 + n;
    float v0 = (k0     < CCH) ? Ua[p * CCH + k0]     : 0.f;
    float v1 = (k0 + 1 < CCH) ? Ua[p * CCH + k0 + 1] : 0.f;
    packB[g] = pack_bf16(v0, v1);
}

// Pre-pack Weff[p,k121] into bf16 WMMA B layout (4 k-steps, K padded to 128)
__global__ void ca_packB2_kernel(const float* __restrict__ Weff, unsigned int* __restrict__ packB) {
    int g = blockIdx.x * blockDim.x + threadIdx.x;   // 4*32*256 threads
    int d  = g & 7;
    int l  = (g >> 3) & 31;
    int nt = (g >> 8) & 31;
    int ks = g >> 13;
    int n = l & 15;
    int half = (l >= 16) ? 8 : 0;
    int kk = ((d < 4) ? (2 * d) : (16 + 2 * (d - 4))) + half;
    int k0 = ks * 32 + kk;
    int p = nt * 16 + n;
    float v0 = (k0     < K2) ? Weff[p * K2 + k0]     : 0.f;
    float v1 = (k0 + 1 < K2) ? Weff[p * K2 + k0 + 1] : 0.f;
    packB[g] = pack_bf16(v0, v1);
}

// ---------------------------------------------------------------------------
// Fused: key GEMM (x·Ua, K=684) + cover im2col GEMM (beta·Weff, K=121)
//        + query/bias add + tanh + Va-dot reduction  -> e[b, hw]
// One workgroup = one (batch, 16-pixel) M-tile; 8 wave32s × 4 N-tiles = NP=512.
// A tiles of the key GEMM are fetched by the Tensor Data Mover (double
// buffered, TENSORcnt-synchronized); B operands are pre-packed bf16 in L2.
__global__ void __launch_bounds__(256)
ca_fused_kernel(const float* __restrict__ x,
                const float* __restrict__ beta,
                const float* __restrict__ qbias,
                const unsigned int* __restrict__ packB1,
                const unsigned int* __restrict__ packB2,
                const float* __restrict__ Va_w,
                const float* __restrict__ Va_b,
                float* __restrict__ e_out) {
    const int b    = blockIdx.y;
    const int m0   = blockIdx.x * 16;             // 16 consecutive hw pixels
    const int t    = threadIdx.x;
    const int lane = t & 31;
    const int wave = t >> 5;

    __shared__ float        tdmBuf[2][32 * 16];   // fp32 A chunks [k][m], 4KB
    __shared__ unsigned int ldsA[256];            // packed 16x32 bf16 A chunk
    __shared__ float        e_part[16];
    if (t < 16) e_part[t] = 0.f;

    v8f acc[4] = {};                              // 4 N-tiles of 16x16 f32 C

    // staging role of this thread: element (M, kk..kk+1) of the A chunk
    const int M  = t & 15;
    const int kp = t >> 4;                        // 0..15
    const int kk = kp * 2;
    int half, dd;
    kslot(kk, half, dd);
    const int dl = M + (half ? 16 : 0);           // dest lane in packed layout
    const unsigned int slot = dl * 8 + dd;

    // ---- key GEMM: A[m,k] = x[b, k, m0+m], K = 684 (pad 704) -------------
#if HAVE_TDM
    const unsigned lds_base = (unsigned)(size_t)&tdmBuf[0][0];
    const float* xtile0 = x + (size_t)b * CCH * HWN + m0;
    if (wave == 0)
        tdm_load_2d(lds_base, xtile0, (unsigned)(HWN - m0), (unsigned)CCH,
                    16u, 32u, (unsigned)HWN);
    for (int ks = 0; ks < KSTEPS1; ++ks) {
        const int buf = ks & 1;
        if (wave == 0) {
            if (ks + 1 < KSTEPS1) {
                // fetch next chunk into the other buffer, then wait for this one
                tdm_load_2d(lds_base + (unsigned)(((ks + 1) & 1) * 2048),
                            xtile0 + (size_t)(ks + 1) * 32 * HWN,
                            (unsigned)(HWN - m0), (unsigned)(CCH - (ks + 1) * 32),
                            16u, 32u, (unsigned)HWN);
                WAIT_TENSORCNT(1);
            } else {
                WAIT_TENSORCNT(0);
            }
        }
        __syncthreads();                          // chunk ready + ldsA reusable
        const float a0 = tdmBuf[buf][kk * 16 + M];        // OOB rows arrive as 0
        const float a1 = tdmBuf[buf][(kk + 1) * 16 + M];
        ldsA[slot] = pack_bf16(a0, a1);
        __syncthreads();
#else
    const size_t xbase = (size_t)b * CCH * HWN + (m0 + M);
    for (int ks = 0; ks < KSTEPS1; ++ks) {
        const int k0 = ks * 32 + kk;
        float a0 = (k0     < CCH) ? x[xbase + (size_t)k0 * HWN]       : 0.f;
        float a1 = (k0 + 1 < CCH) ? x[xbase + (size_t)(k0 + 1) * HWN] : 0.f;
        __syncthreads();
        ldsA[slot] = pack_bf16(a0, a1);
        __syncthreads();
#endif
        ABregs a;
        const u32x4v* lp = (const u32x4v*)ldsA;
        a.u[0] = lp[lane * 2];
        a.u[1] = lp[lane * 2 + 1];

        #pragma unroll
        for (int nt = 0; nt < 4; ++nt) {
            const int ntile = wave * 4 + nt;
            const u32x4v* bp =
                (const u32x4v*)(packB1 + (((ks * NTILES + ntile) * 32 + lane) * 8));
            if (ks + 1 < KSTEPS1)                 // prefetch next k-step's B tile
                __builtin_prefetch(packB1 + ((((ks + 1) * NTILES + ntile) * 32 + lane) * 8), 0, 3);
            ABregs bb; bb.u[0] = bp[0]; bb.u[1] = bp[1];
            acc[nt] = __builtin_amdgcn_wmma_f32_16x16x32_bf16(
                false, a.v, false, bb.v, (short)0, acc[nt], false, false);
        }
    }

    // ---- cover GEMM: A[m,k] = im2col(beta) patch, K = 121 (pad 128) ------
    const int h  = m0 >> 7;                       // W = 128
    const int w0 = m0 & (WW - 1);
    for (int ks = 0; ks < KSTEPS2; ++ks) {
        float a01[2];
        #pragma unroll
        for (int j = 0; j < 2; ++j) {
            const int k0 = ks * 32 + kk + j;
            float v = 0.f;
            if (k0 < K2) {
                const int kh = k0 / KK, kw = k0 - kh * KK;
                const int hh = h + kh - 5;
                const int ww = w0 + M + kw - 5;
                if (hh >= 0 && hh < HH && ww >= 0 && ww < WW)
                    v = beta[b * HWN + hh * WW + ww];
            }
            a01[j] = v;
        }
        __syncthreads();
        ldsA[slot] = pack_bf16(a01[0], a01[1]);
        __syncthreads();

        ABregs a;
        const u32x4v* lp = (const u32x4v*)ldsA;
        a.u[0] = lp[lane * 2];
        a.u[1] = lp[lane * 2 + 1];

        #pragma unroll
        for (int nt = 0; nt < 4; ++nt) {
            const int ntile = wave * 4 + nt;
            const u32x4v* bp =
                (const u32x4v*)(packB2 + (((ks * NTILES + ntile) * 32 + lane) * 8));
            ABregs bb; bb.u[0] = bp[0]; bb.u[1] = bp[1];
            acc[nt] = __builtin_amdgcn_wmma_f32_16x16x32_bf16(
                false, a.v, false, bb.v, (short)0, acc[nt], false, false);
        }
    }

    // ---- epilogue: e[b,m] = sum_p Va[p]*tanh(acc + qbias[b,p]) -----------
    // C layout: lane n = lane&15 is the N (p) column; vgpr i holds row
    // M = i (lanes 0-15) or i+8 (lanes 16-31).
    const int ncol = lane & 15;
    const int rowbase = (lane >= 16) ? 8 : 0;
    #pragma unroll
    for (int nt = 0; nt < 4; ++nt) {
        const int p = (wave * 4 + nt) * 16 + ncol;
        const float q  = qbias[b * NP + p];
        const float va = Va_w[p];
        #pragma unroll
        for (int i = 0; i < 8; ++i) {
            float r = va * tanhf(acc[nt][i] + q);
            r += __shfl_xor(r, 1, 32);
            r += __shfl_xor(r, 2, 32);
            r += __shfl_xor(r, 4, 32);
            r += __shfl_xor(r, 8, 32);            // sum over 16 p-columns in tile
            if (ncol == 0) atomicAdd(&e_part[rowbase + i], r);
        }
    }
    __syncthreads();
    if (t < 16) e_out[b * HWN + m0 + t] = e_part[t] + Va_b[0];
}

// ---------------------------------------------------------------------------
// softmax exactly as the reference: exp / (sum + 1e-8), no max-subtraction.
__global__ void ca_softmax_kernel(const float* __restrict__ e, float* __restrict__ alpha_out) {
    const int b = blockIdx.x;
    const int t = threadIdx.x;
    const int lane = t & 31, wave = t >> 5;
    __shared__ float red[8];
    float v[16];
    float s = 0.f;
    #pragma unroll
    for (int j = 0; j < 16; ++j) {
        v[j] = __expf(e[b * HWN + t + j * 256]);
        s += v[j];
    }
    s += __shfl_xor(s, 1, 32); s += __shfl_xor(s, 2, 32);
    s += __shfl_xor(s, 4, 32); s += __shfl_xor(s, 8, 32); s += __shfl_xor(s, 16, 32);
    if (lane == 0) red[wave] = s;
    __syncthreads();
    float tot = 0.f;
    #pragma unroll
    for (int w = 0; w < 8; ++w) tot += red[w];
    const float inv = 1.f / (tot + 1e-8f);
    #pragma unroll
    for (int j = 0; j < 16; ++j) alpha_out[b * HWN + t + j * 256] = v[j] * inv;
}

// ---------------------------------------------------------------------------
// context[b,c] = sum_hw alpha[b,hw] * x[b,c,hw]   (one wave32 per (b,c) row)
__global__ void ca_context_kernel(const float* __restrict__ x,
                                  const float* __restrict__ alpha,
                                  float* __restrict__ context) {
    const int lane = threadIdx.x & 31;
    const int wave = threadIdx.x >> 5;
    const int row  = blockIdx.x * 8 + wave;       // row = b*C + c, 10944 rows
    const int b = row / CCH;
    const size_t xb = (size_t)row * HWN;
    const float* al = alpha + b * HWN;
    float s = 0.f;
    for (int hw = lane; hw < HWN; hw += 32) s += al[hw] * x[xb + hw];
    s += __shfl_xor(s, 1, 32); s += __shfl_xor(s, 2, 32);
    s += __shfl_xor(s, 4, 32); s += __shfl_xor(s, 8, 32); s += __shfl_xor(s, 16, 32);
    if (lane == 0) context[row] = s;
}

// ---------------------------------------------------------------------------
extern "C" void kernel_launch(void* const* d_in, const int* in_sizes, int n_in,
                              void* d_out, int out_size, void* d_ws, size_t ws_size,
                              hipStream_t stream) {
    const float* x        = (const float*)d_in[0];
    const float* st_hat   = (const float*)d_in[1];
    const float* alpha    = (const float*)d_in[2];
    const float* convQ_w  = (const float*)d_in[3];
    const float* convQ_b  = (const float*)d_in[4];
    const float* Wa_w     = (const float*)d_in[5];
    const float* Wa_b     = (const float*)d_in[6];
    const float* convUa_w = (const float*)d_in[7];
    const float* convUa_b = (const float*)d_in[8];
    const float* Uf_w     = (const float*)d_in[9];
    const float* Uf_b     = (const float*)d_in[10];
    const float* Va_w     = (const float*)d_in[11];
    const float* Va_b     = (const float*)d_in[12];

    float* out_context = (float*)d_out;                    // [16, 684]
    float* out_alpha   = (float*)d_out + BSZ * CCH;        // [16, 4096]

    // workspace layout (bytes)
    char* ws = (char*)d_ws;
    float*        beta   = (float*)(ws + 0);               // 16*4096 f32
    float*        e_buf  = (float*)(ws + 262144);          // 16*4096 f32
    float*        qbias  = (float*)(ws + 524288);          // 16*512  f32
    float*        Weff   = (float*)(ws + 557056);          // 512*121 f32
    unsigned int* packB1 = (unsigned int*)(ws + 804864);   // 22*32*256 u32
    unsigned int* packB2 = (unsigned int*)(ws + 1525760);  // 4*32*256  u32

    ca_beta_kernel  <<<256, 256, 0, stream>>>(alpha, beta);
    ca_weff_kernel  <<<242, 256, 0, stream>>>(Uf_w, convQ_w, Weff);
    ca_qbias_kernel <<< 32, 256, 0, stream>>>(st_hat, Wa_w, Wa_b, convUa_b,
                                              Uf_w, Uf_b, convQ_b, qbias);
    ca_packB1_kernel<<<704, 256, 0, stream>>>(convUa_w, packB1);
    ca_packB2_kernel<<<128, 256, 0, stream>>>(Weff, packB2);

    dim3 grid(HWN / 16, BSZ);
    ca_fused_kernel <<<grid, 256, 0, stream>>>(x, beta, qbias, packB1, packB2,
                                               Va_w, Va_b, e_buf);

    ca_softmax_kernel<<<BSZ, 256, 0, stream>>>(e_buf, out_alpha);
    ca_context_kernel<<<(BSZ * CCH) / 8, 256, 0, stream>>>(x, out_alpha, out_context);
}
